// VSQLayer_19396072308998
// MI455X (gfx1250) — matrix operand order repaired and checked by reading, not compiled, not run on hardware
//
#include <hip/hip_runtime.h>
#include <hip/hip_bf16.h>
#include <stdint.h>

#define B_ 2048
#define T_ 32
#define V_ 1024
#define D_ 256

typedef __attribute__((ext_vector_type(16))) __bf16 v16bf;
typedef __attribute__((ext_vector_type(8)))  float  v8f;
typedef __attribute__((ext_vector_type(4)))  unsigned int v4u;
typedef __attribute__((ext_vector_type(8)))  int v8i_;
typedef __attribute__((ext_vector_type(4)))  int v4i_;

union FragU {
  v16bf v;
  uint4 q[2];
  unsigned int u[8];
  unsigned short h[16];
};

__device__ __forceinline__ unsigned short bf16_rne(float f) {
  unsigned u = __builtin_bit_cast(unsigned, f);
  u += 0x7FFFu + ((u >> 16) & 1u);            // round-to-nearest-even
  return (unsigned short)(u >> 16);
}
__device__ __forceinline__ float bf16_f32(unsigned short h) {
  return __builtin_bit_cast(float, (unsigned)h << 16);
}
__device__ __forceinline__ void split_bf16(float f, unsigned short& hi, unsigned short& lo) {
  hi = bf16_rne(f);
  lo = bf16_rne(f - bf16_f32(hi));
}

// ---- pass 1: c2[t*V+v] = sum_d codebook[t][v][d]^2 ------------------------
__global__ __launch_bounds__(256) void vsq_c2_kernel(const float* __restrict__ cb,
                                                     float* __restrict__ c2buf) {
  const int wave = threadIdx.x >> 5;
  const int lane = threadIdx.x & 31;
  const int row  = blockIdx.x * 8 + wave;       // [0, T_*V_)
  const float* p = cb + (size_t)row * D_ + lane * 8;
  float4 a = *(const float4*)p;
  float4 b = *(const float4*)(p + 4);
  float s = a.x*a.x + a.y*a.y + a.z*a.z + a.w*a.w
          + b.x*b.x + b.y*b.y + b.z*b.z + b.w*b.w;
#pragma unroll
  for (int off = 16; off >= 1; off >>= 1) s += __shfl_xor(s, off, 32);
  if (lane == 0) c2buf[row] = s;
}

// ---- pass 1b: split codebook into bf16 hi/lo planes (one-time) ------------
__global__ __launch_bounds__(256) void vsq_split_kernel(const float* __restrict__ cb,
                                                        unsigned short* __restrict__ cbh,
                                                        unsigned short* __restrict__ cbl) {
  const size_t i = ((size_t)blockIdx.x * 256 + threadIdx.x) * 8;
  float4 a = *(const float4*)(cb + i);
  float4 b = *(const float4*)(cb + i + 4);
  unsigned short h[8], l[8];
  split_bf16(a.x, h[0], l[0]); split_bf16(a.y, h[1], l[1]);
  split_bf16(a.z, h[2], l[2]); split_bf16(a.w, h[3], l[3]);
  split_bf16(b.x, h[4], l[4]); split_bf16(b.y, h[5], l[5]);
  split_bf16(b.z, h[6], l[6]); split_bf16(b.w, h[7], l[7]);
  uint4 hv, lv;
  hv.x = (unsigned)h[0] | ((unsigned)h[1] << 16);
  hv.y = (unsigned)h[2] | ((unsigned)h[3] << 16);
  hv.z = (unsigned)h[4] | ((unsigned)h[5] << 16);
  hv.w = (unsigned)h[6] | ((unsigned)h[7] << 16);
  lv.x = (unsigned)l[0] | ((unsigned)l[1] << 16);
  lv.y = (unsigned)l[2] | ((unsigned)l[3] << 16);
  lv.z = (unsigned)l[4] | ((unsigned)l[5] << 16);
  lv.w = (unsigned)l[6] | ((unsigned)l[7] << 16);
  *(uint4*)(cbh + i) = hv;
  *(uint4*)(cbl + i) = lv;
}

// ---- TDM: load one 16x256 bf16 tile into LDS with 16B/row padding ---------
// D# per ISA ch.8: 2-D tile, data_size=2B, pad_interval=512B, pad_amount=4 dwords
// -> LDS row stride = 132 dwords (bank-conflict-free for 16-lane row reads).
__device__ __forceinline__ void tdm_load_tile(unsigned lds_off,
                                              const unsigned short* gptr) {
  const unsigned long long ga = (unsigned long long)(size_t)gptr;
  v4u g0;
  g0[0] = 1u;                                           // count=1 (valid user D#)
  g0[1] = lds_off;                                      // lds_addr (bytes)
  g0[2] = (unsigned)(ga & 0xFFFFFFFFu);                 // global_addr[31:0]
  g0[3] = (unsigned)((ga >> 32) & 0x1FFFFFFu)           // global_addr[56:32]
        | (2u << 30);                                   // type=2 ("image")
  v8i_ g1;
  g1[0] = (int)((1u << 16)        // data_size = 1 -> 2 bytes
              | (1u << 20)        // pad_enable
              | (6u << 22)        // pad_interval: code 6 -> every 128 dwords (512B)
              | (3u << 25));      // pad_amount:   code 3 -> 4 dwords (16B)
  g1[1] = (int)(256u << 16);      // tensor_dim0 = 256 (bits 79:48, low half)
  g1[2] = (int)(16u << 16);       // tensor_dim0 hi = 0 ; tensor_dim1 = 16 (low half)
  g1[3] = (int)(256u << 16);      // tensor_dim1 hi = 0 ; tile_dim0 = 256
  g1[4] = (int)16u;               // tile_dim1 = 16 ; tile_dim2 = 0
  g1[5] = (int)256;               // tensor_dim0_stride[31:0] = 256 elements
  g1[6] = 0;                      // stride0 hi ; tensor_dim1_stride lo
  g1[7] = 0;
  const v4i_ z4 = {0, 0, 0, 0};
  const v8i_ z8 = {0, 0, 0, 0, 0, 0, 0, 0};
  __builtin_amdgcn_tensor_load_to_lds(g0, g1, z4, z4, z8, 0);
}

__device__ __forceinline__ unsigned lds_off_of(const void* p) {
  return (unsigned)(size_t)p;     // LDS aperture: addr[31:0] is the LDS byte offset
}

// ---- pass 2 (TDM path): WMMA distance search + argmin + gather ------------
__global__ __launch_bounds__(256) void vsq_main_tdm(const float* __restrict__ xin,
                                                    const unsigned short* __restrict__ cbh,
                                                    const unsigned short* __restrict__ cbl,
                                                    const float* __restrict__ cb,
                                                    const float* __restrict__ c2buf,
                                                    float* __restrict__ embed_out,
                                                    int* __restrict__ idx_out) {
  __shared__ unsigned int s_b[2][2][16 * 132];  // [buf][hi/lo][row-stride 132 dw]
  __shared__ int s_bestv[128];

  const int tid  = threadIdx.x;
  const int wave = tid >> 5;
  const int lane = tid & 31;
  const int t      = blockIdx.y;
  const int b_base = blockIdx.x * 128;
  const float* cbt = cb + (size_t)t * V_ * D_;
  const unsigned short* cbh_t = cbh + (size_t)t * V_ * D_;
  const unsigned short* cbl_t = cbl + (size_t)t * V_ * D_;

  // A fragments: 16 rows x 256 K, ISA 16-bit A layout (one-time per block).
  const int m  = lane & 15;
  const int ak = (lane >> 4) * 8;
  const float* xrow = xin + ((size_t)(b_base + wave * 16 + m) * T_ + t) * D_;
  FragU ahi[8], alo[8];
#pragma unroll
  for (int kk = 0; kk < 8; ++kk) {
    const float* p = xrow + kk * 32 + ak;
    float4 c0 = *(const float4*)(p);
    float4 c1 = *(const float4*)(p + 4);
    float4 c2v = *(const float4*)(p + 16);
    float4 c3 = *(const float4*)(p + 20);
    float fe[16] = {c0.x,c0.y,c0.z,c0.w, c1.x,c1.y,c1.z,c1.w,
                    c2v.x,c2v.y,c2v.z,c2v.w, c3.x,c3.y,c3.z,c3.w};
#pragma unroll
    for (int e = 0; e < 16; ++e) split_bf16(fe[e], ahi[kk].h[e], alo[kk].h[e]);
  }

  float bd[8];
  int   bv[8];
#pragma unroll
  for (int i = 0; i < 8; ++i) { bd[i] = 3.402823466e38f; bv[i] = 0; }

  const int n   = lane & 15;
  const int bko = (lane >> 4) * 4;

  // prologue: wave 0 prefetches tile 0 via the Tensor Data Mover
  if (wave == 0) {
    tdm_load_tile(lds_off_of(&s_b[0][0][0]), cbh_t);
    tdm_load_tile(lds_off_of(&s_b[0][1][0]), cbl_t);
  }

  for (int it = 0; it < V_ / 16; ++it) {
    const int buf = it & 1;
    __syncthreads();                     // all waves done reading s_b[buf^1]
    if (wave == 0) {
      if (it + 1 < V_ / 16) {
        const size_t off = (size_t)(it + 1) * 16 * D_;
        tdm_load_tile(lds_off_of(&s_b[buf ^ 1][0][0]), cbh_t + off);
        tdm_load_tile(lds_off_of(&s_b[buf ^ 1][1][0]), cbl_t + off);
        __builtin_amdgcn_s_wait_tensorcnt((short)2);  // tile `it` complete (in-order)
      } else {
        __builtin_amdgcn_s_wait_tensorcnt((short)0);
      }
    }
    __syncthreads();                     // s_b[buf] visible to all waves

    v8f acc = {};
#pragma unroll
    for (int kk = 0; kk < 8; ++kk) {
      FragU bh, bl;
      const int db = n * 132 + kk * 16 + bko;
      bh.q[0] = *(const uint4*)&s_b[buf][0][db];
      bh.q[1] = *(const uint4*)&s_b[buf][0][db + 8];
      bl.q[0] = *(const uint4*)&s_b[buf][1][db];
      bl.q[1] = *(const uint4*)&s_b[buf][1][db + 8];
      // bf16x3: hi*hi + hi*lo + lo*hi (f32 accumulate)
      acc = __builtin_amdgcn_wmma_f32_16x16x32_bf16(false, ahi[kk].v, false, bh.v,
                                                    (short)0, acc, false, false);
      acc = __builtin_amdgcn_wmma_f32_16x16x32_bf16(false, ahi[kk].v, false, bl.v,
                                                    (short)0, acc, false, false);
      acc = __builtin_amdgcn_wmma_f32_16x16x32_bf16(false, alo[kk].v, false, bh.v,
                                                    (short)0, acc, false, false);
    }

    const float c2n  = c2buf[t * V_ + it * 16 + n];
    const int   vidx = it * 16 + n;
#pragma unroll
    for (int i = 0; i < 8; ++i) {
      const float d2 = c2n - 2.0f * acc[i];
      if (d2 < bd[i]) { bd[i] = d2; bv[i] = vidx; }
    }
  }

  // argmin across the 16 lanes of each half-wave (first-min tie-break)
#pragma unroll
  for (int off = 8; off >= 1; off >>= 1) {
#pragma unroll
    for (int i = 0; i < 8; ++i) {
      const float od = __shfl_xor(bd[i], off, 32);
      const int   ov = __shfl_xor(bv[i], off, 32);
      if (od < bd[i] || (od == bd[i] && ov < bv[i])) { bd[i] = od; bv[i] = ov; }
    }
  }

  if (lane == 0 || lane == 16) {
    const int rbase = wave * 16 + (lane >> 4) * 8;
#pragma unroll
    for (int i = 0; i < 8; ++i) {
      s_bestv[rbase + i] = bv[i];
      idx_out[(size_t)(b_base + rbase + i) * T_ + t] = bv[i];
    }
  }
  __syncthreads();

  for (int p = tid; p < 128 * 64; p += 256) {
    const int row = p >> 6, q = p & 63;
    const int v = s_bestv[row];
    const float4 f = *(const float4*)(cbt + (size_t)v * D_ + q * 4);
    *(float4*)(embed_out + ((size_t)(b_base + row) * T_ + t) * D_ + q * 4) = f;
  }
}

// ---- pass 2 (fallback, small workspace): in-kernel conversion -------------
__global__ __launch_bounds__(256) void vsq_main_fallback(const float* __restrict__ xin,
                                                         const float* __restrict__ cb,
                                                         const float* __restrict__ c2buf,
                                                         float* __restrict__ embed_out,
                                                         int* __restrict__ idx_out) {
  __shared__ unsigned int s_hi[16 * 132];
  __shared__ unsigned int s_lo[16 * 132];
  __shared__ int s_bestv[128];

  const int tid  = threadIdx.x;
  const int wave = tid >> 5;
  const int lane = tid & 31;
  const int t      = blockIdx.y;
  const int b_base = blockIdx.x * 128;
  const float* cbt = cb + (size_t)t * V_ * D_;

  const int m  = lane & 15;
  const int ak = (lane >> 4) * 8;
  const float* xrow = xin + ((size_t)(b_base + wave * 16 + m) * T_ + t) * D_;
  FragU ahi[8], alo[8];
#pragma unroll
  for (int kk = 0; kk < 8; ++kk) {
    const float* p = xrow + kk * 32 + ak;
    float4 c0 = *(const float4*)(p);
    float4 c1 = *(const float4*)(p + 4);
    float4 c2v = *(const float4*)(p + 16);
    float4 c3 = *(const float4*)(p + 20);
    float fe[16] = {c0.x,c0.y,c0.z,c0.w, c1.x,c1.y,c1.z,c1.w,
                    c2v.x,c2v.y,c2v.z,c2v.w, c3.x,c3.y,c3.z,c3.w};
#pragma unroll
    for (int e = 0; e < 16; ++e) split_bf16(fe[e], ahi[kk].h[e], alo[kk].h[e]);
  }

  float bd[8];
  int   bv[8];
#pragma unroll
  for (int i = 0; i < 8; ++i) { bd[i] = 3.402823466e38f; bv[i] = 0; }

  const int n   = lane & 15;
  const int bko = (lane >> 4) * 4;

  for (int v0 = 0; v0 < V_; v0 += 16) {
    __syncthreads();
#pragma unroll
    for (int j = 0; j < 4; ++j) {
      int p = tid + j * 256;
      int row = p >> 6, col4 = p & 63;
      const float4 f = *(const float4*)(cbt + (size_t)(v0 + row) * D_ + col4 * 4);
      unsigned short h0,h1,h2,h3,l0,l1,l2,l3;
      split_bf16(f.x, h0, l0); split_bf16(f.y, h1, l1);
      split_bf16(f.z, h2, l2); split_bf16(f.w, h3, l3);
      uint2 hw, lw;
      hw.x = (unsigned)h0 | ((unsigned)h1 << 16);
      hw.y = (unsigned)h2 | ((unsigned)h3 << 16);
      lw.x = (unsigned)l0 | ((unsigned)l1 << 16);
      lw.y = (unsigned)l2 | ((unsigned)l3 << 16);
      *(uint2*)&s_hi[row * 132 + col4 * 2] = hw;
      *(uint2*)&s_lo[row * 132 + col4 * 2] = lw;
    }
    __syncthreads();

    v8f acc = {};
#pragma unroll
    for (int kk = 0; kk < 8; ++kk) {
      FragU bh, bl;
      const int db = n * 132 + kk * 16 + bko;
      bh.q[0] = *(const uint4*)&s_hi[db];
      bh.q[1] = *(const uint4*)&s_hi[db + 8];
      bl.q[0] = *(const uint4*)&s_lo[db];
      bl.q[1] = *(const uint4*)&s_lo[db + 8];
      acc = __builtin_amdgcn_wmma_f32_16x16x32_bf16(false, ahi[kk].v, false, bh.v,
                                                    (short)0, acc, false, false);
      acc = __builtin_amdgcn_wmma_f32_16x16x32_bf16(false, ahi[kk].v, false, bl.v,
                                                    (short)0, acc, false, false);
      acc = __builtin_amdgcn_wmma_f32_16x16x32_bf16(false, alo[kk].v, false, bh.v,
                                                    (short)0, acc, false, false);
    }

    const float c2n  = c2buf[t * V_ + v0 + n];
    const int   vidx = v0 + n;
#pragma unroll
    for (int i = 0; i < 8; ++i) {
      const float d2 = c2n - 2.0f * acc[i];
      if (d2 < bd[i]) { bd[i] = d2; bv[i] = vidx; }
    }
  }

#pragma unroll
  for (int off = 8; off >= 1; off >>= 1) {
#pragma unroll
    for (int i = 0; i < 8; ++i) {
      const float od = __shfl_xor(bd[i], off, 32);
      const int   ov = __shfl_xor(bv[i], off, 32);
      if (od < bd[i] || (od == bd[i] && ov < bv[i])) { bd[i] = od; bv[i] = ov; }
    }
  }

  if (lane == 0 || lane == 16) {
    const int rbase = wave * 16 + (lane >> 4) * 8;
#pragma unroll
    for (int i = 0; i < 8; ++i) {
      s_bestv[rbase + i] = bv[i];
      idx_out[(size_t)(b_base + rbase + i) * T_ + t] = bv[i];
    }
  }
  __syncthreads();

  for (int p = tid; p < 128 * 64; p += 256) {
    const int row = p >> 6, q = p & 63;
    const int v = s_bestv[row];
    const float4 f = *(const float4*)(cbt + (size_t)v * D_ + q * 4);
    *(float4*)(embed_out + ((size_t)(b_base + row) * T_ + t) * D_ + q * 4) = f;
  }
}

extern "C" void kernel_launch(void* const* d_in, const int* in_sizes, int n_in,
                              void* d_out, int out_size, void* d_ws, size_t ws_size,
                              hipStream_t stream) {
  const float* xin = (const float*)d_in[0];   // (B,T,D) f32
  const float* cb  = (const float*)d_in[1];   // (T,V,D) f32
  float* embed_out = (float*)d_out;                         // (B,T,D) f32
  int*   idx_out   = (int*)d_out + (size_t)B_ * T_ * D_;    // (B,T) i32 bits

  const size_t plane   = (size_t)T_ * V_ * D_;              // bf16 elements per plane
  const size_t c2bytes = (size_t)T_ * V_ * sizeof(float);   // 128 KB
  const size_t need    = c2bytes + 2 * plane * sizeof(unsigned short); // ~32.1 MB

  float* c2buf = (float*)d_ws;
  vsq_c2_kernel<<<dim3(T_ * V_ / 8), dim3(256), 0, stream>>>(cb, c2buf);

  if (ws_size >= need) {
    unsigned short* cbh = (unsigned short*)((char*)d_ws + c2bytes);
    unsigned short* cbl = cbh + plane;
    vsq_split_kernel<<<dim3((unsigned)(plane / 2048)), dim3(256), 0, stream>>>(cb, cbh, cbl);
    vsq_main_tdm<<<dim3(B_ / 128, T_), dim3(256), 0, stream>>>(xin, cbh, cbl, cb,
                                                               c2buf, embed_out, idx_out);
  } else {
    vsq_main_fallback<<<dim3(B_ / 128, T_), dim3(256), 0, stream>>>(xin, cb, c2buf,
                                                                    embed_out, idx_out);
  }
}
